// AdvancedNeuralStructureAnalyzer_3204045603438
// MI455X (gfx1250) — matrix operand order repaired and checked by reading, not compiled, not run on hardware
//
#include <hip/hip_runtime.h>
#include <hip/hip_bf16.h>

typedef __bf16 bf16;
typedef bf16 v16bf __attribute__((ext_vector_type(16)));
typedef float v8f  __attribute__((ext_vector_type(8)));

#define BB 2
#define NN 256
#define DD 768
#define EE 8192
#define HH 8
#define HDD 96
#define NPAIR 32640
#define CPAIR 8160
#define NCHUNK 4

static __host__ int cdiv(int a, int b) { return (a + b - 1) / b; }

static __device__ __forceinline__ v8f wmma_bf16(const v16bf& a, const v16bf& b, const v8f& c) {
    return __builtin_amdgcn_wmma_f32_16x16x32_bf16(false, a, false, b, (short)0, c, false, false);
}

// ---------------------------------------------------------------------------
// Big bf16 WMMA GEMM: C[M,Nc] = alpha * (A @ opB) + bias, opt ReLU.
//   TRANSB=1: B is (Nc x K) row-major, C = A * B^T   (linear-layer weights)
//   TRANSB=0: B is (K x Nc) row-major, C = A * B
// Requirements: M % 64 == 0, K % 32 == 0, Nc % 32 == 0.
// One 32x32 macro-tile per wave (2 M-tiles x 2 N-tiles, 4 accumulators,
// 4 WMMAs per k-step with 2x reuse of both operand fragments). Because
// Nc % 32 == 0, both N-tiles of a wave are valid together -> a single
// wave-uniform early return, and an unguarded branch-free inner loop
// (EXEC stays all-ones around WMMA as the ISA requires).
// Block = 128 threads (4 waves) -> 64x64 block tile.
// Batched via grid.z: off = (z/nz1)*s2 + (z%nz1)*s1 (element offsets).
// Writes f32 C and/or bf16 Cb (non-null ones), same ldc.
// ---------------------------------------------------------------------------
template<int TRANSB, int RELU>
__global__ void gemm_wmma_big(
    const bf16* __restrict__ A,  long sA1, long sA2, int lda,
    const bf16* __restrict__ Bm, long sB1, long sB2, int ldb,
    float* __restrict__ C, bf16* __restrict__ Cb, long sC1, long sC2, int ldc,
    const float* __restrict__ bias, float alpha,
    int M, int Nc, int K, int nz1)
{
    const int z  = blockIdx.z;
    const int zb = z / nz1;
    const int zh = z % nz1;
    A  += (long)zb * sA2 + (long)zh * sA1;
    Bm += (long)zb * sB2 + (long)zh * sB1;
    const long coff = (long)zb * sC2 + (long)zh * sC1;

    const int wave  = threadIdx.x >> 5;
    const int lane  = threadIdx.x & 31;
    const int lhalf = lane >> 4;     // 0 or 1
    const int l16   = lane & 15;

    const int tn0 = (blockIdx.x * 2 + (wave & 1)) * 2;   // first of 2 n-tiles
    const int tm0 = (blockIdx.y * 2 + (wave >> 1)) * 2;  // first of 2 m-tiles
    const int ntiles = Nc >> 4;
    if (tn0 >= ntiles) return;       // wave-uniform; Nc%32==0 -> pair valid together

    // A rows / B cols owned by this lane
    const int am0 = tm0 * 16 + l16;
    const int am1 = am0 + 16;
    const int bn0 = tn0 * 16 + l16;
    const int bn1 = bn0 + 16;

    const bf16* arow0 = A + (long)am0 * lda;
    const bf16* arow1 = A + (long)am1 * lda;

    v8f acc00 = {}, acc01 = {}, acc10 = {}, acc11 = {};
    for (int k0 = 0; k0 < K; k0 += 32) {
        // A fragment: lanes 0-15 -> K {0..15}; lanes 16-31 -> K {16..31}
        v16bf a0, a1;
        {
            const bf16* ap0 = arow0 + k0 + lhalf * 16;
            const bf16* ap1 = arow1 + k0 + lhalf * 16;
            #pragma unroll
            for (int i = 0; i < 8; ++i) {
                a0[i] = ap0[i];   a0[8 + i] = ap0[8 + i];
                a1[i] = ap1[i];   a1[8 + i] = ap1[8 + i];
            }
        }
        // B fragment: lane holds 16 consecutive K of its column (lanes 16-31 -> K+16)
        const int klo = k0 + lhalf * 16;
        v16bf b0, b1;
        if (TRANSB) {
            const bf16* bp0 = Bm + (long)bn0 * ldb + klo;
            const bf16* bp1 = Bm + (long)bn1 * ldb + klo;
            #pragma unroll
            for (int i = 0; i < 16; ++i) { b0[i] = bp0[i]; b1[i] = bp1[i]; }
        } else {
            const bf16* bp = Bm + (long)klo * ldb;
            #pragma unroll
            for (int i = 0; i < 16; ++i) {
                b0[i] = bp[(long)i * ldb + bn0];
                b1[i] = bp[(long)i * ldb + bn1];
            }
        }
        acc00 = wmma_bf16(a0, b0, acc00);
        acc10 = wmma_bf16(a1, b0, acc10);
        acc01 = wmma_bf16(a0, b1, acc01);
        acc11 = wmma_bf16(a1, b1, acc11);
    }

    const float bv0 = bias ? bias[bn0] : 0.0f;
    const float bv1 = bias ? bias[bn1] : 0.0f;
    #pragma unroll
    for (int r = 0; r < 8; ++r) {
        const int mr0 = tm0 * 16 + r + lhalf * 8;   // C layout: vgpr r -> M=r (+8 hi half)
        const int mr1 = mr0 + 16;
        float v00 = acc00[r] * alpha + bv0;
        float v10 = acc10[r] * alpha + bv0;
        float v01 = acc01[r] * alpha + bv1;
        float v11 = acc11[r] * alpha + bv1;
        if (RELU) {
            v00 = fmaxf(v00, 0.f); v10 = fmaxf(v10, 0.f);
            v01 = fmaxf(v01, 0.f); v11 = fmaxf(v11, 0.f);
        }
        const long o00 = coff + (long)mr0 * ldc + bn0;
        const long o10 = coff + (long)mr1 * ldc + bn0;
        const long o01 = coff + (long)mr0 * ldc + bn1;
        const long o11 = coff + (long)mr1 * ldc + bn1;
        if (C)  { C[o00] = v00;  C[o10] = v10;  C[o01] = v01;  C[o11] = v11; }
        if (Cb) { Cb[o00] = (bf16)v00; Cb[o10] = (bf16)v10;
                  Cb[o01] = (bf16)v01; Cb[o11] = (bf16)v11; }
    }
}

// ---------------------------------------------------------------------------
// Small guarded WMMA GEMM for padded narrow outputs (Nc = 8 / 1).
// One 16x16 tile per wave; per-lane column guard.
// ---------------------------------------------------------------------------
template<int RELU>
__global__ void gemm_wmma_small(
    const bf16* __restrict__ A, int lda,
    const bf16* __restrict__ Bm, int ldb,
    float* __restrict__ C, int ldc,
    const float* __restrict__ bias,
    int M, int Nc, int K)
{
    const int wave = threadIdx.x >> 5;
    const int lane = threadIdx.x & 31;
    const int tm = blockIdx.y * 4 + wave;      // 4 waves stacked along M
    if (tm >= (M >> 4)) return;

    const int lhalf = lane >> 4;
    const int l16   = lane & 15;
    const int am = tm * 16 + l16;
    const int bn = l16;                        // single N tile
    const bool bnok = (bn < Nc);

    v8f acc = {};
    for (int k0 = 0; k0 < K; k0 += 32) {
        v16bf a;
        {
            const bf16* ap = A + (long)am * lda + k0 + lhalf * 16;
            #pragma unroll
            for (int i = 0; i < 8; ++i) { a[i] = ap[i]; a[8 + i] = ap[8 + i]; }
        }
        v16bf b;
        const int klo = k0 + lhalf * 16;
        if (bnok) {
            const bf16* bp = Bm + (long)bn * ldb + klo;
            #pragma unroll
            for (int i = 0; i < 16; ++i) b[i] = bp[i];
        } else {
            #pragma unroll
            for (int i = 0; i < 16; ++i) b[i] = (bf16)0.0f;
        }
        acc = wmma_bf16(a, b, acc);
    }
    if (!bnok) return;
    const float bv = bias ? bias[bn] : 0.0f;
    #pragma unroll
    for (int r = 0; r < 8; ++r) {
        const int mrow = tm * 16 + r + lhalf * 8;
        float v = acc[r] + bv;
        if (RELU) v = fmaxf(v, 0.0f);
        C[(long)mrow * ldc + bn] = v;
    }
}

// ---------------------------------------------------------------------------
// Elementwise / reduction kernels
// ---------------------------------------------------------------------------
__global__ void k_f32_to_bf16(const float* __restrict__ in, bf16* __restrict__ out, long n) {
    long i = (long)blockIdx.x * blockDim.x + threadIdx.x;
    if (i < n) out[i] = (bf16)in[i];
}

__global__ void k_zero_f32(float* __restrict__ p, long n) {
    long i = (long)blockIdx.x * blockDim.x + threadIdx.x;
    if (i < n) p[i] = 0.0f;
}

// agg[:, tgt[e], :] += w[e] * g[:, src[e], :]   (both batches per thread)
__global__ void k_scatter_add(const float* __restrict__ g, const float* __restrict__ ew,
                              const int* __restrict__ ei, float* __restrict__ agg) {
    long idx = (long)blockIdx.x * blockDim.x + threadIdx.x;
    if (idx >= (long)EE * DD) return;
    int e = (int)(idx / DD), d = (int)(idx % DD);
    int s = ei[e * 2], t = ei[e * 2 + 1];
    float w = ew[e];
    const long BS = (long)NN * DD;
    atomicAdd(&agg[(long)t * DD + d],      w * g[(long)s * DD + d]);
    atomicAdd(&agg[BS + (long)t * DD + d], w * g[BS + (long)s * DD + d]);
}

// o = relu(self+neigh); layernorm over D; write f32 + bf16. One block per row.
__global__ void k_add_relu_ln(const float* __restrict__ sf, const float* __restrict__ nf,
                              const float* __restrict__ gamma, const float* __restrict__ beta,
                              float* __restrict__ gout, bf16* __restrict__ goutb) {
    const int row = blockIdx.x;
    const long base = (long)row * DD;
    __shared__ float s_sum[256];
    __shared__ float s_sq[256];
    float ls = 0.f, lq = 0.f;
    for (int d = threadIdx.x; d < DD; d += 256) {
        float v = fmaxf(sf[base + d] + nf[base + d], 0.f);
        ls += v; lq += v * v;
    }
    s_sum[threadIdx.x] = ls; s_sq[threadIdx.x] = lq;
    __syncthreads();
    for (int s = 128; s > 0; s >>= 1) {
        if (threadIdx.x < s) {
            s_sum[threadIdx.x] += s_sum[threadIdx.x + s];
            s_sq[threadIdx.x]  += s_sq[threadIdx.x + s];
        }
        __syncthreads();
    }
    const float mu  = s_sum[0] / DD;
    const float var = s_sq[0] / DD - mu * mu;
    const float inv = rsqrtf(var + 1e-5f);
    for (int d = threadIdx.x; d < DD; d += 256) {
        float v = fmaxf(sf[base + d] + nf[base + d], 0.f);
        float o = gamma[d] * (v - mu) * inv + beta[d];
        gout[base + d]  = o;
        goutb[base + d] = (bf16)o;
    }
}

// softmax over 256 cols, f32 in-place + bf16 copy. One block (256 thr) per row.
__global__ void k_softmax_attn(float* __restrict__ x, bf16* __restrict__ xb) {
    const long base = (long)blockIdx.x * NN;
    __shared__ float red[256];
    float v = x[base + threadIdx.x];
    red[threadIdx.x] = v;
    __syncthreads();
    for (int s = 128; s > 0; s >>= 1) {
        if (threadIdx.x < s) red[threadIdx.x] = fmaxf(red[threadIdx.x], red[threadIdx.x + s]);
        __syncthreads();
    }
    const float m = red[0];
    __syncthreads();
    float e = expf(v - m);
    red[threadIdx.x] = e;
    __syncthreads();
    for (int s = 128; s > 0; s >>= 1) {
        if (threadIdx.x < s) red[threadIdx.x] += red[threadIdx.x + s];
        __syncthreads();
    }
    const float a = e / red[0];
    x[base + threadIdx.x]  = a;
    xb[base + threadIdx.x] = (bf16)a;
}

__global__ void k_mean_heads(const float* __restrict__ attn, float* __restrict__ aw) {
    int idx = blockIdx.x * blockDim.x + threadIdx.x;
    if (idx >= BB * NN * NN) return;
    int k = idx % NN, q = (idx / NN) % NN, b = idx / (NN * NN);
    float s = 0.f;
    #pragma unroll
    for (int h = 0; h < HH; ++h)
        s += attn[(((long)(b * HH + h)) * NN + q) * NN + k];
    aw[idx] = s * 0.125f;
}

// 8-wide softmax from padded (ld=16) logits to contiguous 8-wide output
__global__ void k_softmax8(const float* __restrict__ in, float* __restrict__ out, int rows) {
    int r = blockIdx.x * blockDim.x + threadIdx.x;
    if (r >= rows) return;
    const float* p = in + (long)r * 16;
    float m = p[0];
    #pragma unroll
    for (int i = 1; i < 8; ++i) m = fmaxf(m, p[i]);
    float e[8], s = 0.f;
    #pragma unroll
    for (int i = 0; i < 8; ++i) { e[i] = expf(p[i] - m); s += e[i]; }
    const float inv = 1.f / s;
    #pragma unroll
    for (int i = 0; i < 8; ++i) out[(long)r * 8 + i] = e[i] * inv;
}

// relation chunk softmax: row r in [0, 2*CP): b=r/CP, p = chunk0 + r%CP
__global__ void k_softmax8_rel(const float* __restrict__ in, float* __restrict__ out,
                               int CP, int chunk0) {
    int r = blockIdx.x * blockDim.x + threadIdx.x;
    if (r >= 2 * CP) return;
    int b = r / CP, pl = r % CP;
    const float* p = in + (long)r * 16;
    float m = p[0];
    #pragma unroll
    for (int i = 1; i < 8; ++i) m = fmaxf(m, p[i]);
    float e[8], s = 0.f;
    #pragma unroll
    for (int i = 0; i < 8; ++i) { e[i] = expf(p[i] - m); s += e[i]; }
    const float inv = 1.f / s;
    float* q = out + ((long)b * NPAIR + chunk0 + pl) * 8;
    #pragma unroll
    for (int i = 0; i < 8; ++i) q[i] = e[i] * inv;
}

__global__ void k_sigmoid(const float* __restrict__ in, float* __restrict__ out, int rows) {
    int r = blockIdx.x * blockDim.x + threadIdx.x;
    if (r >= rows) return;
    out[r] = 1.f / (1.f + expf(-in[(long)r * 16]));
}

__global__ void k_make_pairs(int* __restrict__ ii, int* __restrict__ jj) {
    int i = threadIdx.x;
    if (i >= NN) return;
    int off = i * (NN - 1) - i * (i - 1) / 2;
    for (int j = i + 1; j < NN; ++j) { ii[off] = i; jj[off] = j; ++off; }
}

// h1[b, pl, k] = relu(P[b, ii[p], k] + Q[b, jj[p], k] + b1[k]) as bf16
__global__ void k_gather_h1(const float* __restrict__ P, const float* __restrict__ Q,
                            const float* __restrict__ b1, const int* __restrict__ ii,
                            const int* __restrict__ jj, bf16* __restrict__ h1b,
                            int chunk0, int CP) {
    long idx = (long)blockIdx.x * blockDim.x + threadIdx.x;
    const long total = (long)2 * CP * DD;
    if (idx >= total) return;
    int k  = (int)(idx % DD);
    long t = idx / DD;
    int pl = (int)(t % CP);
    int b  = (int)(t / CP);
    int p  = chunk0 + pl;
    int i = ii[p], j = jj[p];
    float v = P[((long)b * NN + i) * DD + k] + Q[((long)b * NN + j) * DD + k] + b1[k];
    h1b[idx] = (bf16)fmaxf(v, 0.f);
}

// ---------------------------------------------------------------------------
// Host-side launch helpers
// ---------------------------------------------------------------------------
static void launch_gemm(hipStream_t s, int transB, int relu,
                        const bf16* A, long sA1, long sA2, int lda,
                        const bf16* Bm, long sB1, long sB2, int ldb,
                        float* C, bf16* Cb, long sC1, long sC2, int ldc,
                        const float* bias, float alpha,
                        int M, int Nc, int K, int nz, int nz1) {
    // M % 64 == 0, K % 32 == 0, Nc % 32 == 0 required
    dim3 grid(cdiv(Nc / 32, 2), (M / 16) / 4, nz);
    dim3 block(128);
    if (transB) {
        if (relu) gemm_wmma_big<1, 1><<<grid, block, 0, s>>>(A, sA1, sA2, lda, Bm, sB1, sB2, ldb, C, Cb, sC1, sC2, ldc, bias, alpha, M, Nc, K, nz1);
        else      gemm_wmma_big<1, 0><<<grid, block, 0, s>>>(A, sA1, sA2, lda, Bm, sB1, sB2, ldb, C, Cb, sC1, sC2, ldc, bias, alpha, M, Nc, K, nz1);
    } else {
        if (relu) gemm_wmma_big<0, 1><<<grid, block, 0, s>>>(A, sA1, sA2, lda, Bm, sB1, sB2, ldb, C, Cb, sC1, sC2, ldc, bias, alpha, M, Nc, K, nz1);
        else      gemm_wmma_big<0, 0><<<grid, block, 0, s>>>(A, sA1, sA2, lda, Bm, sB1, sB2, ldb, C, Cb, sC1, sC2, ldc, bias, alpha, M, Nc, K, nz1);
    }
}

static void launch_gemm_small(hipStream_t s, const bf16* A, int lda, const bf16* Bm, int ldb,
                              float* C, int ldc, const float* bias, int M, int Nc, int K) {
    dim3 grid(1, cdiv(M / 16, 4), 1);
    dim3 block(128);
    gemm_wmma_small<0><<<grid, block, 0, s>>>(A, lda, Bm, ldb, C, ldc, bias, M, Nc, K);
}

static void launch_conv(hipStream_t s, const float* in, bf16* out, long n) {
    k_f32_to_bf16<<<(int)((n + 255) / 256), 256, 0, s>>>(in, out, n);
}

extern "C" void kernel_launch(void* const* d_in, const int* in_sizes, int n_in,
                              void* d_out, int out_size, void* d_ws, size_t ws_size,
                              hipStream_t stream) {
    (void)in_sizes; (void)n_in; (void)out_size; (void)ws_size;
    const float* x   = (const float*)d_in[0];
    const float* ew  = (const float*)d_in[1];
    const int*   ei  = (const int*)d_in[2];
    const float* oc_b1 = (const float*)d_in[4];
    const float* oc_b2 = (const float*)d_in[6];
    const float* oc_b3 = (const float*)d_in[8];
    const float* ip_b1 = (const float*)d_in[10];
    const float* ip_b2 = (const float*)d_in[12];
    const float* ip_b3 = (const float*)d_in[14];
    const float* rc_b1 = (const float*)d_in[16];
    const float* rc_b2 = (const float*)d_in[18];
    const float* rc_b3 = (const float*)d_in[20];
    const float* ai_b  = (const float*)d_in[22];
    const float* ao_b  = (const float*)d_in[24];

    float* out = (float*)d_out;
    const long OC_OFF = 0, REL_OFF = 4096, IMP_OFF = 526336, ATT_OFF = 526848, AW_OFF = 920064;

    // bump allocator on workspace
    char* wp = (char*)d_ws;
    auto alloc = [&](size_t bytes) -> void* {
        void* p = (void*)wp;
        wp += (bytes + 255) & ~(size_t)255;
        return p;
    };
    const long MB = (long)BB * NN;           // 512 rows
    const long XSZ = MB * DD;                // 393216

    bf16* xb    = (bf16*)alloc(XSZ * 2);
    bf16* woc1b = (bf16*)alloc((size_t)DD * DD * 2);
    bf16* woc2b = (bf16*)alloc((size_t)(DD / 2) * DD * 2);
    bf16* woc3b = (bf16*)alloc((size_t)8 * (DD / 2) * 2);
    bf16* wip1b = (bf16*)alloc((size_t)(DD / 2) * DD * 2);
    bf16* wip2b = (bf16*)alloc((size_t)(DD / 4) * (DD / 2) * 2);
    bf16* wip3b = (bf16*)alloc((size_t)(DD / 4) * 2);
    bf16* wrc1b = (bf16*)alloc((size_t)DD * 2 * DD * 2);
    bf16* wrc2b = (bf16*)alloc((size_t)(DD / 2) * DD * 2);
    bf16* wrc3b = (bf16*)alloc((size_t)8 * (DD / 2) * 2);
    bf16* waib  = (bf16*)alloc((size_t)3 * DD * DD * 2);
    bf16* waob  = (bf16*)alloc((size_t)DD * DD * 2);
    bf16* wgsb[3], *wgnb[3];
    for (int l = 0; l < 3; ++l) {
        wgsb[l] = (bf16*)alloc((size_t)DD * DD * 2);
        wgnb[l] = (bf16*)alloc((size_t)DD * DD * 2);
    }
    bf16* t1b = (bf16*)alloc(XSZ * 2);
    bf16* t2b = (bf16*)alloc(MB * (DD / 2) * 2);
    bf16* u1b = (bf16*)alloc(MB * (DD / 2) * 2);
    bf16* u2b = (bf16*)alloc(MB * (DD / 4) * 2);
    bf16* aggb = (bf16*)alloc(XSZ * 2);
    bf16* gbA = (bf16*)alloc(XSZ * 2);
    bf16* gbB = (bf16*)alloc(XSZ * 2);
    bf16* qkvb = (bf16*)alloc(MB * 3 * DD * 2);
    bf16* attnb = (bf16*)alloc((size_t)BB * HH * NN * NN * 2);
    bf16* ctxb = (bf16*)alloc(XSZ * 2);
    bf16* attb = (bf16*)alloc(XSZ * 2);
    bf16* h1b  = (bf16*)alloc((size_t)2 * CPAIR * DD * 2);
    bf16* h2b  = (bf16*)alloc((size_t)2 * CPAIR * (DD / 2) * 2);

    float* locf  = (float*)alloc(MB * 16 * 4);
    float* uimpf = (float*)alloc(MB * 16 * 4);
    float* selff = (float*)alloc(XSZ * 4);
    float* neighf = (float*)alloc(XSZ * 4);
    float* agg   = (float*)alloc(XSZ * 4);
    float* gfA   = (float*)alloc(XSZ * 4);
    float* gfB   = (float*)alloc(XSZ * 4);
    float* scores = (float*)alloc((size_t)BB * HH * NN * NN * 4);
    float* Pf    = (float*)alloc(XSZ * 4);
    float* Qf    = (float*)alloc(XSZ * 4);
    float* l3f   = (float*)alloc((size_t)2 * CPAIR * 16 * 4);
    int*   iiv   = (int*)alloc(NPAIR * 4);
    int*   jjv   = (int*)alloc(NPAIR * 4);

    // ---- convert inputs & weights to bf16 ----
    launch_conv(stream, x, xb, XSZ);
    launch_conv(stream, (const float*)d_in[3],  woc1b, (long)DD * DD);
    launch_conv(stream, (const float*)d_in[5],  woc2b, (long)(DD / 2) * DD);
    launch_conv(stream, (const float*)d_in[7],  woc3b, (long)8 * (DD / 2));
    launch_conv(stream, (const float*)d_in[9],  wip1b, (long)(DD / 2) * DD);
    launch_conv(stream, (const float*)d_in[11], wip2b, (long)(DD / 4) * (DD / 2));
    launch_conv(stream, (const float*)d_in[13], wip3b, (long)(DD / 4));
    launch_conv(stream, (const float*)d_in[15], wrc1b, (long)DD * 2 * DD);
    launch_conv(stream, (const float*)d_in[17], wrc2b, (long)(DD / 2) * DD);
    launch_conv(stream, (const float*)d_in[19], wrc3b, (long)8 * (DD / 2));
    launch_conv(stream, (const float*)d_in[21], waib,  (long)3 * DD * DD);
    launch_conv(stream, (const float*)d_in[23], waob,  (long)DD * DD);
    for (int l = 0; l < 3; ++l) {
        launch_conv(stream, (const float*)d_in[25 + 6 * l + 0], wgsb[l], (long)DD * DD);
        launch_conv(stream, (const float*)d_in[25 + 6 * l + 2], wgnb[l], (long)DD * DD);
    }

    // ---- object classifier: relu(xW1)->relu(W2)->softmax(W3) ----
    launch_gemm(stream, 1, 1, xb, 0, 0, DD, woc1b, 0, 0, DD, nullptr, t1b, 0, 0, DD, oc_b1, 1.f, MB, DD, DD, 1, 1);
    launch_gemm(stream, 1, 1, t1b, 0, 0, DD, woc2b, 0, 0, DD, nullptr, t2b, 0, 0, DD / 2, oc_b2, 1.f, MB, DD / 2, DD, 1, 1);
    launch_gemm_small(stream, t2b, DD / 2, woc3b, DD / 2, locf, 16, oc_b3, MB, 8, DD / 2);
    k_softmax8<<<cdiv(MB, 256), 256, 0, stream>>>(locf, out + OC_OFF, MB);

    // ---- importance: relu->relu->sigmoid ----
    launch_gemm(stream, 1, 1, xb, 0, 0, DD, wip1b, 0, 0, DD, nullptr, u1b, 0, 0, DD / 2, ip_b1, 1.f, MB, DD / 2, DD, 1, 1);
    launch_gemm(stream, 1, 1, u1b, 0, 0, DD / 2, wip2b, 0, 0, DD / 2, nullptr, u2b, 0, 0, DD / 4, ip_b2, 1.f, MB, DD / 4, DD / 2, 1, 1);
    launch_gemm_small(stream, u2b, DD / 4, wip3b, DD / 4, uimpf, 16, ip_b3, MB, 1, DD / 4);
    k_sigmoid<<<cdiv(MB, 256), 256, 0, stream>>>(uimpf, out + IMP_OFF, MB);

    // ---- GNN: 3 layers of self + scatter-add neighbor + LN ----
    const float* gcur = x;
    const bf16*  gcurb = xb;
    float* gf_next[3] = { gfA, gfB, gfA };
    bf16*  gb_next[3] = { gbA, gbB, gbA };
    for (int l = 0; l < 3; ++l) {
        const float* bs = (const float*)d_in[25 + 6 * l + 1];
        const float* bn = (const float*)d_in[25 + 6 * l + 3];
        const float* gm = (const float*)d_in[25 + 6 * l + 4];
        const float* bt = (const float*)d_in[25 + 6 * l + 5];
        launch_gemm(stream, 1, 0, gcurb, 0, 0, DD, wgsb[l], 0, 0, DD, selff, nullptr, 0, 0, DD, bs, 1.f, MB, DD, DD, 1, 1);
        k_zero_f32<<<(int)((XSZ + 255) / 256), 256, 0, stream>>>(agg, XSZ);
        k_scatter_add<<<(int)(((long)EE * DD + 255) / 256), 256, 0, stream>>>(gcur, ew, ei, agg);
        launch_conv(stream, agg, aggb, XSZ);
        launch_gemm(stream, 1, 0, aggb, 0, 0, DD, wgnb[l], 0, 0, DD, neighf, nullptr, 0, 0, DD, bn, 1.f, MB, DD, DD, 1, 1);
        k_add_relu_ln<<<(int)MB, 256, 0, stream>>>(selff, neighf, gm, bt, gf_next[l], gb_next[l]);
        gcur = gf_next[l];
        gcurb = gb_next[l];
    }

    // ---- attention ----
    launch_gemm(stream, 1, 0, gcurb, 0, 0, DD, waib, 0, 0, DD, nullptr, qkvb, 0, 0, 3 * DD, ai_b, 1.f, MB, 3 * DD, DD, 1, 1);
    const long QKV_B = (long)NN * 3 * DD;   // per-batch stride in qkv
    const long SC_H  = (long)NN * NN;       // per-head stride in scores
    // scores[b,h] = q @ k^T / sqrt(HD)
    launch_gemm(stream, 1, 0,
                qkvb,        HDD, QKV_B, 3 * DD,
                qkvb + DD,   HDD, QKV_B, 3 * DD,
                scores, nullptr, SC_H, SC_H * HH, NN,
                nullptr, 0.1020620726159658f, NN, NN, HDD, BB * HH, HH);
    k_softmax_attn<<<BB * HH * NN, 256, 0, stream>>>(scores, attnb);
    k_mean_heads<<<cdiv(BB * NN * NN, 256), 256, 0, stream>>>(scores, out + AW_OFF);
    // ctx[b,h] = attn @ v  (non-transposed B)
    launch_gemm(stream, 0, 0,
                attnb,           SC_H, SC_H * HH, NN,
                qkvb + 2 * DD,   HDD,  QKV_B,     3 * DD,
                nullptr, ctxb, HDD, (long)NN * DD, DD,
                nullptr, 1.f, NN, HDD, NN, BB * HH, HH);
    // attended = ctx @ ao_w^T + b  -> f32 into d_out, bf16 copy for relation
    launch_gemm(stream, 1, 0, ctxb, 0, 0, DD, waob, 0, 0, DD, out + ATT_OFF, attb, 0, 0, DD, ao_b, 1.f, MB, DD, DD, 1, 1);

    // ---- relation: split rc_w1 into halves; P = att*W1a^T, Q = att*W1b^T ----
    launch_gemm(stream, 1, 0, attb, 0, 0, DD, wrc1b,      0, 0, 2 * DD, Pf, nullptr, 0, 0, DD, nullptr, 1.f, MB, DD, DD, 1, 1);
    launch_gemm(stream, 1, 0, attb, 0, 0, DD, wrc1b + DD, 0, 0, 2 * DD, Qf, nullptr, 0, 0, DD, nullptr, 1.f, MB, DD, DD, 1, 1);
    k_make_pairs<<<1, 256, 0, stream>>>(iiv, jjv);
    for (int c = 0; c < NCHUNK; ++c) {
        const long h1n = (long)2 * CPAIR * DD;
        k_gather_h1<<<(int)((h1n + 255) / 256), 256, 0, stream>>>(Pf, Qf, rc_b1, iiv, jjv, h1b, c * CPAIR, CPAIR);
        launch_gemm(stream, 1, 1, h1b, 0, 0, DD, wrc2b, 0, 0, DD, nullptr, h2b, 0, 0, DD / 2, rc_b2, 1.f, 2 * CPAIR, DD / 2, DD, 1, 1);
        launch_gemm_small(stream, h2b, DD / 2, wrc3b, DD / 2, l3f, 16, rc_b3, 2 * CPAIR, 8, DD / 2);
        k_softmax8_rel<<<cdiv(2 * CPAIR, 256), 256, 0, stream>>>(l3f, out + REL_OFF, CPAIR, c * CPAIR);
    }
}
